// MoETaskAttention_18528488915483
// MI455X (gfx1250) — matrix operand order repaired and checked by reading, not compiled, not run on hardware
//
#include <hip/hip_runtime.h>
#include <hip/hip_bf16.h>
#include <stdint.h>

// ---------------------------------------------------------------------------
// MoE task-attention for MI455X (gfx1250, wave32, WMMA).
// All heavy math goes through v_wmma_f32_16x16x32_bf16 (fp32 accumulate).
// Compute-bound workload (~85 GFLOP, ~90MB working set -> L2 resident).
// ---------------------------------------------------------------------------

typedef __attribute__((ext_vector_type(16))) __bf16 v16bf;
typedef __attribute__((ext_vector_type(8)))  float  v8f;

#define TOK    8192      // B*N
#define C_DIM  768
#define E_NUM  24
#define H_NUM  8
#define HD_DIM 96
#define EH     2304      // E_NUM * HD_DIM
#define KV_N   192       // 2*HD
#define ATT_SCALE 0.10206207261596575f   // 96^-0.5

static __device__ __forceinline__ __bf16 f2bf(float f) {
  uint32_t u = __builtin_bit_cast(uint32_t, f);
  u = u + 0x7FFFu + ((u >> 16) & 1u);            // round-to-nearest-even
  uint16_t s = (uint16_t)(u >> 16);
  return __builtin_bit_cast(__bf16, s);
}

// A-fragment (16x32 bf16, row-major source). Lane r (=lane&15) holds row r;
// half h (=lane>>4): K = h*16+j (contiguous, matches packed B).
static __device__ __forceinline__ v16bf load_a_frag(const __bf16* base, int lda, int lane) {
  int r = lane & 15, hh = lane >> 4;
  const __bf16* p = base + (size_t)r * lda + hh * 16;
  v16bf a;
#pragma unroll
  for (int j = 0; j < 8; ++j) { a[j] = p[j]; a[8 + j] = p[8 + j]; }
  return a;
}

// B-fragment from K-block-packed weights: packed[(k/32)*N*32 + n*32 + (k&31)]
// -> per lane two contiguous 16B loads.  Lane col = lane&15, K = h*16+j.
static __device__ __forceinline__ v16bf load_b_packed(const __bf16* Bp, int Ntot,
                                                      int kb, int n0, int lane) {
  int c = lane & 15, hh = lane >> 4;
  const __bf16* p = Bp + ((size_t)kb * Ntot + (size_t)(n0 + c)) * 32 + hh * 16;
  v16bf b;
#pragma unroll
  for (int j = 0; j < 16; ++j) b[j] = p[j];
  return b;
}

static __device__ __forceinline__ v8f wmma_bf16(v16bf a, v16bf b, v8f c) {
  return __builtin_amdgcn_wmma_f32_16x16x32_bf16(false, a, false, b, (short)0, c,
                                                 false, false);
}

// ---------------------------------------------------------------------------
// Pack fp32 weights -> bf16, K-blocked fragment-native layout.
// mode 0: src row-major (K x N).  mode 1: W_in (E,C,HD) with k=c, n=e*96+d.
// ---------------------------------------------------------------------------
__global__ void pack_b_kernel(const float* __restrict__ src, __bf16* __restrict__ dst,
                              int K, int Ntot, int mode) {
  int idx = blockIdx.x * blockDim.x + threadIdx.x;
  if (idx >= K * Ntot) return;
  int k = idx / Ntot, n = idx - k * Ntot;
  float v;
  if (mode == 0) v = src[(size_t)k * Ntot + n];
  else { int e = n / HD_DIM, d = n - e * HD_DIM;
         v = src[((size_t)e * C_DIM + k) * HD_DIM + d]; }
  dst[((size_t)(k >> 5) * Ntot + n) * 32 + (k & 31)] = f2bf(v);
}

__global__ void zero_aux_kernel(float* aux) {
  if (threadIdx.x < 64) aux[threadIdx.x] = 0.f;
}

// ---------------------------------------------------------------------------
// Gating: one wave per token. Fused x->bf16 convert, logits, softmax, lse,
// register top-8 (bitmask selection: no dynamic local stores -> no scratch),
// atomic importance/load/zloss accumulation.
// ---------------------------------------------------------------------------
__global__ __launch_bounds__(256) void gate_kernel(
    const float* __restrict__ x, const float* __restrict__ Wg_all,
    const int* __restrict__ task_p, __bf16* __restrict__ xb,
    int* __restrict__ topi, float* __restrict__ topg, float* __restrict__ aux) {
  int lane = threadIdx.x & 31, wave = threadIdx.x >> 5;
  int tok = blockIdx.x * 8 + wave;
  const float* Wg = Wg_all + (size_t)(*task_p) * C_DIM * E_NUM;
  const float* xr = x + (size_t)tok * C_DIM;
  float acc[E_NUM];
#pragma unroll
  for (int e = 0; e < E_NUM; ++e) acc[e] = 0.f;
  for (int c = lane; c < C_DIM; c += 32) {
    float xv = xr[c];
    xb[(size_t)tok * C_DIM + c] = f2bf(xv);
    const float* wr = Wg + (size_t)c * E_NUM;
#pragma unroll
    for (int e = 0; e < E_NUM; ++e) acc[e] = fmaf(xv, wr[e], acc[e]);
  }
#pragma unroll
  for (int e = 0; e < E_NUM; ++e)
#pragma unroll
    for (int m = 16; m >= 1; m >>= 1) acc[e] += __shfl_xor(acc[e], m, 32);

  float mx = acc[0];
#pragma unroll
  for (int e = 1; e < E_NUM; ++e) mx = fmaxf(mx, acc[e]);
  float se = 0.f, pr[E_NUM];
#pragma unroll
  for (int e = 0; e < E_NUM; ++e) { pr[e] = __expf(acc[e] - mx); se += pr[e]; }
  float inv = 1.f / se;
#pragma unroll
  for (int e = 0; e < E_NUM; ++e) pr[e] *= inv;
  float lse = mx + __logf(se);

  unsigned chosen = 0u;
  int selE[H_NUM]; float selV[H_NUM]; float gsum = 0.f;
#pragma unroll
  for (int h = 0; h < H_NUM; ++h) {
    float bv = -1.f; int be = 0;
#pragma unroll
    for (int e = 0; e < E_NUM; ++e) {
      bool ok = !((chosen >> e) & 1u) && (pr[e] > bv);
      bv = ok ? pr[e] : bv; be = ok ? e : be;
    }
    chosen |= (1u << be);
    selE[h] = be; selV[h] = bv; gsum += bv;
  }
  float gn = 1.f / (gsum + 1e-6f);
  if (lane == 0) {
#pragma unroll
    for (int h = 0; h < H_NUM; ++h) {
      topi[tok * H_NUM + h] = selE[h];
      topg[tok * H_NUM + h] = selV[h] * gn;
      atomicAdd(&aux[E_NUM + selE[h]], 1.f);
    }
#pragma unroll
    for (int e = 0; e < E_NUM; ++e) atomicAdd(&aux[e], pr[e]);
    atomicAdd(&aux[2 * E_NUM], lse * lse);
  }
}

// ---------------------------------------------------------------------------
// Generic bf16 WMMA GEMM: D(MxN) = A(MxK,row-major) @ Bp(K-blocked packed).
// Block = 8 waves, tile 256x64; wave computes 32x64 (2 A-frags reuse each
// B-frag for 2 WMMAs -> 8 WMMAs per 32-K step, ~21 FLOP/B of issued loads).
// Speculative global_prefetch of the A tile two K-steps ahead.
// ---------------------------------------------------------------------------
template <bool OUTF32>
__global__ __launch_bounds__(256) void gemm_bf16_kernel(
    const __bf16* __restrict__ A, const __bf16* __restrict__ Bp,
    void* __restrict__ D, int Ntot, int K, int lda, int ldd) {
  int lane = threadIdx.x & 31, wave = threadIdx.x >> 5;
  int row0 = blockIdx.y * 256 + wave * 32;
  int col0 = blockIdx.x * 64;
  v8f acc[2][4];
#pragma unroll
  for (int m = 0; m < 2; ++m)
#pragma unroll
    for (int t = 0; t < 4; ++t)
#pragma unroll
      for (int i = 0; i < 8; ++i) acc[m][t][i] = 0.f;

  int r = lane & 15;
  for (int k0 = 0; k0 < K; k0 += 32) {
    if (k0 + 64 < K) {   // speculative prefetch, two K-steps ahead
      __builtin_prefetch(A + (size_t)(row0 + r) * lda + k0 + 64, 0, 3);
      __builtin_prefetch(A + (size_t)(row0 + 16 + r) * lda + k0 + 64, 0, 3);
    }
    v16bf a0 = load_a_frag(A + (size_t)row0 * lda + k0, lda, lane);
    v16bf a1 = load_a_frag(A + (size_t)(row0 + 16) * lda + k0, lda, lane);
#pragma unroll
    for (int t = 0; t < 4; ++t) {
      v16bf b = load_b_packed(Bp, Ntot, k0 >> 5, col0 + t * 16, lane);
      acc[0][t] = wmma_bf16(a0, b, acc[0][t]);
      acc[1][t] = wmma_bf16(a1, b, acc[1][t]);
    }
  }
  int cl = lane & 15, hh = lane >> 4;
#pragma unroll
  for (int m = 0; m < 2; ++m)
#pragma unroll
    for (int t = 0; t < 4; ++t)
#pragma unroll
      for (int i = 0; i < 8; ++i) {
        size_t row = row0 + m * 16 + hh * 8 + i;
        size_t col = col0 + t * 16 + cl;
        if constexpr (OUTF32) ((float*)D)[row * ldd + col] = acc[m][t][i];
        else                  ((__bf16*)D)[row * ldd + col] = f2bf(acc[m][t][i]);
      }
}

// ---------------------------------------------------------------------------
// Flash attention: one wave per (b, h, 16-row i-tile). j in steps of 32:
// 6 WMMAs (QK^T for two 16-col tiles) + online softmax + 6 WMMAs (P@V).
// P is transposed C-layout -> A-layout through a per-wave LDS tile.
// Q rows gathered through top_i (per-row expert indirection into qall).
// ---------------------------------------------------------------------------
__global__ __launch_bounds__(256) void attn_kernel(
    const __bf16* __restrict__ qall, const __bf16* __restrict__ kvb,
    const int* __restrict__ topi, float* __restrict__ ob) {
  __shared__ __bf16 plds[8][16][36];   // padded to dodge bank conflicts
  int lane = threadIdx.x & 31, wave = threadIdx.x >> 5;
  int b = blockIdx.y >> 3, h = blockIdx.y & 7;
  int i0 = (blockIdx.x * 8 + wave) * 16;
  int cl = lane & 15, hh = lane >> 4, r = cl;

  // Q fragments (row r uses its own routed expert)
  int rtok = b * 1024 + i0 + r;
  int e = topi[rtok * H_NUM + h];
  const __bf16* qbase = qall + (size_t)rtok * EH + (size_t)e * HD_DIM;
  v16bf qf[3];
#pragma unroll
  for (int kk = 0; kk < 3; ++kk) {
    const __bf16* p = qbase + kk * 32 + hh * 16;
#pragma unroll
    for (int j = 0; j < 8; ++j) { qf[kk][j] = p[j]; qf[kk][8 + j] = p[8 + j]; }
  }

  v8f o[6]; float mrow[8], lrow[8];
#pragma unroll
  for (int t = 0; t < 6; ++t)
#pragma unroll
    for (int i = 0; i < 8; ++i) o[t][i] = 0.f;
#pragma unroll
  for (int i = 0; i < 8; ++i) { mrow[i] = -1e30f; lrow[i] = 0.f; }

  const __bf16* kvbB = kvb + (size_t)b * 1024 * KV_N;

  for (int jt = 0; jt < 1024; jt += 32) {
    if (jt + 32 < 1024) {   // speculative prefetch of next K/V j-block
      __builtin_prefetch(kvbB + (size_t)(jt + 32 + cl) * KV_N, 0, 3);
      __builtin_prefetch(kvbB + (size_t)(jt + 48 + cl) * KV_N, 0, 3);
    }
    v8f s0, s1;
#pragma unroll
    for (int i = 0; i < 8; ++i) { s0[i] = 0.f; s1[i] = 0.f; }
#pragma unroll
    for (int kk = 0; kk < 3; ++kk) {
      v16bf bk0, bk1;  // B[k][n] = K[jt+n][kk*32 + k] -> contiguous per lane
      const __bf16* p0 = kvbB + (size_t)(jt + cl) * KV_N + kk * 32 + hh * 16;
      const __bf16* p1 = kvbB + (size_t)(jt + 16 + cl) * KV_N + kk * 32 + hh * 16;
#pragma unroll
      for (int j = 0; j < 16; ++j) { bk0[j] = p0[j]; bk1[j] = p1[j]; }
      s0 = wmma_bf16(qf[kk], bk0, s0);
      s1 = wmma_bf16(qf[kk], bk1, s1);
    }
    float alpha[8];
#pragma unroll
    for (int i = 0; i < 8; ++i) {
      s0[i] *= ATT_SCALE; s1[i] *= ATT_SCALE;
      float cmax = fmaxf(s0[i], s1[i]);
#pragma unroll
      for (int m = 8; m >= 1; m >>= 1) cmax = fmaxf(cmax, __shfl_xor(cmax, m, 32));
      float mnew = fmaxf(mrow[i], cmax);
      float p0v = __expf(s0[i] - mnew);
      float p1v = __expf(s1[i] - mnew);
      float rs = p0v + p1v;
#pragma unroll
      for (int m = 8; m >= 1; m >>= 1) rs += __shfl_xor(rs, m, 32);
      alpha[i] = __expf(mrow[i] - mnew);
      lrow[i] = lrow[i] * alpha[i] + rs;
      mrow[i] = mnew;
      s0[i] = p0v; s1[i] = p1v;
    }
    __syncthreads();   // protect previous iteration's LDS reads
#pragma unroll
    for (int i = 0; i < 8; ++i) {
      plds[wave][hh * 8 + i][cl]      = f2bf(s0[i]);
      plds[wave][hh * 8 + i][16 + cl] = f2bf(s1[i]);
    }
    __syncthreads();
    v16bf pa;   // P as A-fragment (K = column of P)
#pragma unroll
    for (int j = 0; j < 8; ++j) {
      pa[j]     = plds[wave][r][hh * 16 + j];
      pa[8 + j] = plds[wave][r][hh * 16 + 8 + j];
    }
#pragma unroll
    for (int t = 0; t < 6; ++t) {
#pragma unroll
      for (int i = 0; i < 8; ++i) o[t][i] *= alpha[i];
      v16bf bv;   // B[k][n] = V[jt+k][t*16+n]
      const __bf16* pv = kvbB + (size_t)(jt + hh * 16) * KV_N + HD_DIM + t * 16 + cl;
#pragma unroll
      for (int j = 0; j < 16; ++j) bv[j] = pv[(size_t)j * KV_N];
      o[t] = wmma_bf16(pa, bv, o[t]);
    }
  }
#pragma unroll
  for (int t = 0; t < 6; ++t)
#pragma unroll
    for (int i = 0; i < 8; ++i) {
      int row = hh * 8 + i;
      size_t tok = (size_t)b * 1024 + i0 + row;
      ob[(tok * H_NUM + h) * HD_DIM + t * 16 + cl] = o[t][i] / lrow[i];
    }
}

// ---------------------------------------------------------------------------
// Scatter gated attention output into dense z (B*N x E*HD) for the final GEMM.
// ---------------------------------------------------------------------------
__global__ __launch_bounds__(256) void scatter_z_kernel(
    const float* __restrict__ ob, const int* __restrict__ topi,
    const float* __restrict__ topg, __bf16* __restrict__ z) {
  __shared__ int slot[E_NUM];
  __shared__ float gg[H_NUM];
  int tok = blockIdx.x, t = threadIdx.x;
  if (t < E_NUM) slot[t] = -1;
  __syncthreads();
  if (t < H_NUM) { slot[topi[tok * H_NUM + t]] = t; gg[t] = topg[tok * H_NUM + t]; }
  __syncthreads();
  for (int idx = t; idx < EH; idx += 256) {
    int e = idx / HD_DIM, d = idx - e * HD_DIM;
    int h = slot[e];
    float v = 0.f;
    if (h >= 0) v = gg[h] * ob[((size_t)tok * H_NUM + h) * HD_DIM + d];
    z[(size_t)tok * EH + idx] = f2bf(v);
  }
}

__global__ void aux_final_kernel(const float* __restrict__ aux, float* out_aux) {
  float sw = 0.f;
  for (int e = 0; e < E_NUM; ++e) {
    float imp = aux[e] / (float)TOK;
    float ld  = aux[E_NUM + e] / (float)(TOK * H_NUM);
    sw += imp * ld;
  }
  float switchloss = (float)E_NUM * sw;
  float zloss = aux[2 * E_NUM] / (float)TOK;
  *out_aux = 0.1f * switchloss + 0.001f * zloss;
}

// ---------------------------------------------------------------------------
extern "C" void kernel_launch(void* const* d_in, const int* in_sizes, int n_in,
                              void* d_out, int out_size, void* d_ws, size_t ws_size,
                              hipStream_t stream) {
  (void)in_sizes; (void)n_in; (void)out_size; (void)ws_size;
  const float* x      = (const float*)d_in[0];
  const float* W_gate = (const float*)d_in[1];
  const float* W_in   = (const float*)d_in[2];
  const float* W_out  = (const float*)d_in[3];
  const float* W_kv   = (const float*)d_in[4];
  const int*   task   = (const int*)d_in[5];

  char* ws = (char*)d_ws; size_t off = 0;
  auto alloc = [&](size_t bytes) { void* p = ws + off;
                                   off = (off + bytes + 255) & ~(size_t)255; return p; };
  __bf16* xb    = (__bf16*)alloc((size_t)TOK * C_DIM * 2);   // 12.6 MB
  __bf16* qall  = (__bf16*)alloc((size_t)TOK * EH * 2);      // 37.7 MB (reused as z)
  __bf16* kvb   = (__bf16*)alloc((size_t)TOK * KV_N * 2);    //  3.1 MB
  float*  ob    = (float*)alloc((size_t)TOK * H_NUM * HD_DIM * 4);  // 25.2 MB
  __bf16* winp  = (__bf16*)alloc((size_t)C_DIM * EH * 2);    //  3.5 MB
  __bf16* woutp = (__bf16*)alloc((size_t)EH * C_DIM * 2);    //  3.5 MB
  __bf16* wkvp  = (__bf16*)alloc((size_t)C_DIM * KV_N * 2);  //  0.3 MB
  int*    topi  = (int*)alloc((size_t)TOK * H_NUM * 4);
  float*  topg  = (float*)alloc((size_t)TOK * H_NUM * 4);
  float*  aux   = (float*)alloc(256);                        // total ~87 MB

  zero_aux_kernel<<<1, 64, 0, stream>>>(aux);
  pack_b_kernel<<<(C_DIM * KV_N + 255) / 256, 256, 0, stream>>>(W_kv, wkvp, C_DIM, KV_N, 0);
  pack_b_kernel<<<(C_DIM * EH + 255) / 256, 256, 0, stream>>>(W_in, winp, C_DIM, EH, 1);
  pack_b_kernel<<<(EH * C_DIM + 255) / 256, 256, 0, stream>>>(W_out, woutp, EH, C_DIM, 0);
  gate_kernel<<<TOK / 8, 256, 0, stream>>>(x, W_gate, task, xb, topi, topg, aux);
  // q_all = X @ W_in  (8192 x 2304, K=768)
  gemm_bf16_kernel<false><<<dim3(EH / 64, TOK / 256), 256, 0, stream>>>(
      xb, winp, qall, EH, C_DIM, C_DIM, EH);
  // kv = X @ W_kv  (8192 x 192, K=768)
  gemm_bf16_kernel<false><<<dim3(KV_N / 64, TOK / 256), 256, 0, stream>>>(
      xb, wkvp, kvb, KV_N, C_DIM, C_DIM, KV_N);
  attn_kernel<<<dim3(8, 64), 256, 0, stream>>>(qall, kvb, topi, ob);
  scatter_z_kernel<<<TOK, 256, 0, stream>>>(ob, topi, topg, qall /* z aliases qall */);
  // out = Z @ W_out  (8192 x 768, K=2304), fp32 straight into d_out
  gemm_bf16_kernel<true><<<dim3(C_DIM / 64, TOK / 256), 256, 0, stream>>>(
      qall, woutp, d_out, C_DIM, EH, EH, C_DIM);
  aux_final_kernel<<<1, 1, 0, stream>>>(aux, (float*)d_out + (size_t)TOK * C_DIM);
}
